// GraphMambaKo_5162550689851
// MI455X (gfx1250) — matrix operand order repaired and checked by reading, not compiled, not run on hardware
//
#include <hip/hip_runtime.h>
#include <hip/hip_bf16.h>

// ---------------------------------------------------------------------------
// GraphMamba on MI455X (gfx1250).
// enc = broadcast(feat) over (b,l) and h0 = state_init for all b => every (b,l) row of
// every mamba step is identical. The model collapses to: encoder GEMM
// (121x3072 . 3072x8) -> tiny MLP -> adapter+LN -> single-row 7-step
// recurrence -> broadcast 6 vectors of 121 floats to (6,128,24,121).
// Roofline: ~1.5MB read + ~8.9MB write => ~0.45us at 23.3 TB/s.
//
// Matrix ops: v_wmma_f32_16x16x32_f16 with B operands pre-packed into the
// exact 32x16 f16 fragment layout (ISA 7.12.2). Encoder x tiles are staged
// into LDS by the Tensor Data Mover (tensor_load_to_lds + s_wait_tensorcnt);
// the builtin arity differs per toolchain (5-arg ROCm 7.2 vs 6-arg
// amdgpu-toolchain, detected via the TDM header), with a float4 copy fallback.
// ---------------------------------------------------------------------------

typedef __attribute__((ext_vector_type(16))) _Float16 v16h;
typedef __attribute__((ext_vector_type(8)))  float    v8f;
typedef __attribute__((ext_vector_type(4)))  unsigned int v4u;
typedef __attribute__((ext_vector_type(8)))  int      v8i;
typedef __attribute__((ext_vector_type(4)))  int      v4i;

#define NNODES 121
#define DMODEL 121
#define SSTATE 64
#define HID    128
#define BL     3072          // 128*24
#define STEPS  6

// TDM dispatch: 0 = fallback copy, 1 = 5-arg builtin (ROCm 7.2),
// 2 = 6-arg builtin (amdgpu-toolchain, ships the TDM header).
#define TDM_NONE 0
#define TDM_5ARG 1
#define TDM_6ARG 2
#if __has_builtin(__builtin_amdgcn_tensor_load_to_lds)
# if __has_include(<hip/amd_detail/amd_gfx1250_TDM.h>)
#  define TDM_MODE TDM_6ARG
# else
#  define TDM_MODE TDM_5ARG
# endif
#else
# define TDM_MODE TDM_NONE
#endif

// Packed-fragment tile bases (tiles of 32 lanes x 16 f16 = 32B/lane).
#define TB_LIFT1 0     // 4x8
#define TB_LIFT2 32    // 4x8
#define TB_NET1  64    // 8x8
#define TB_NET2  128   // 4x8
#define TB_FC1   160   // 4x8
#define TB_FC2   192   // 4x4
#define TB_FC3   208   // 4x4
#define TB_FCOUT 224   // 4x8
#define TB_CONV  256   // 96x1
#define NTILES   352

#define ACT_NONE 0
#define ACT_RELU 1
#define ACT_TANH 2
#define ACT_SOFTPLUS 3

// ---- A fragment (16x32 f16) from padded LDS vector: row m==0 = vector ------
__device__ __forceinline__ v16h frag_a_rowvec(const float* vin, int kt) {
  int lane = threadIdx.x & 31;
  int m = lane & 15, half = lane >> 4;
  float sel = (m == 0) ? 1.f : 0.f;
  v16h a;
#pragma unroll
  for (int r = 0; r < 8; ++r) {
    int kp = (r < 4) ? (2 * r + half * 8) : (16 + 2 * (r - 4) + half * 8);
    int k = kt * 32 + kp;
    a[2 * r]     = (_Float16)(vin[k]     * sel);
    a[2 * r + 1] = (_Float16)(vin[k + 1] * sel);
  }
  return a;
}

// vout[n] = act( vin(1xK) . W(KxN) + bias ), B from packed fragments.
__device__ __forceinline__ void vecmat_wmma_pk(const float* vin,
                                               const v16h* __restrict__ pk,
                                               int tileBase, int ncols,
                                               const float* __restrict__ bias,
                                               float* vout, int KT, int NT,
                                               int act) {
  int tid = threadIdx.x, lane = tid & 31, wave = tid >> 5;
  for (int nt = wave; nt < NT; nt += 8) {
    v8f c = {};
    for (int kt = 0; kt < KT; ++kt) {
      v16h a = frag_a_rowvec(vin, kt);
      v16h b = pk[(tileBase + nt * KT + kt) * 32 + lane];  // 32B vector load
      c = __builtin_amdgcn_wmma_f32_16x16x32_f16(false, a, false, b,
                                                 (short)0, c, false, false);
    }
    if (lane < 16) {                 // row 0 of D: lanes 0-15, vgpr 0
      int n = nt * 16 + lane;
      if (n < ncols) {
        float y = c[0] + bias[n];
        if      (act == ACT_RELU)     y = fmaxf(y, 0.f);
        else if (act == ACT_TANH)     y = tanhf(y);
        else if (act == ACT_SOFTPLUS) y = log1pf(__expf(y));
        vout[n] = y;
      }
    }
  }
}

// ---- Kernel 0: pack all weight matrices into B-fragment layout -------------
__global__ __launch_bounds__(256) void k0_pack(
    const float* __restrict__ lift_w1, const float* __restrict__ lift_w2,
    const float* __restrict__ net_w1,  const float* __restrict__ net_w2,
    const float* __restrict__ fc1_w,   const float* __restrict__ fc2_w,
    const float* __restrict__ fc3_w,   const float* __restrict__ fcout_w,
    const float* __restrict__ conv_w,  v16h* __restrict__ pk) {
  int gid = blockIdx.x * 256 + threadIdx.x;
  int tile = gid >> 5, lane = gid & 31;
  if (tile >= NTILES) return;

  int t = tile;
  const float* W; int krows, ncols, KT;
  if      (t < TB_LIFT2) {            W = lift_w1; krows = 121;  ncols = 128; KT = 4; }
  else if (t < TB_NET1)  { t -= TB_LIFT2; W = lift_w2; krows = 128;  ncols = 121; KT = 4; }
  else if (t < TB_NET2)  { t -= TB_NET1;  W = net_w1;  krows = 242;  ncols = 128; KT = 8; }
  else if (t < TB_FC1)   { t -= TB_NET2;  W = net_w2;  krows = 128;  ncols = 121; KT = 4; }
  else if (t < TB_FC2)   { t -= TB_FC1;   W = fc1_w;   krows = 121;  ncols = 121; KT = 4; }
  else if (t < TB_FC3)   { t -= TB_FC2;   W = fc2_w;   krows = 121;  ncols = 64;  KT = 4; }
  else if (t < TB_FCOUT) { t -= TB_FC3;   W = fc3_w;   krows = 121;  ncols = 64;  KT = 4; }
  else if (t < TB_CONV)  { t -= TB_FCOUT; W = fcout_w; krows = 121;  ncols = 121; KT = 4; }
  else                   { t -= TB_CONV;  W = conv_w;  krows = 3072; ncols = 8;   KT = 96; }
  int kt = t % KT, nt = t / KT;

  int half = lane >> 4;
  int n  = nt * 16 + (lane & 15);
  int nc = (n < ncols) ? n : (ncols - 1);          // clamp: load stays in bounds
  float nm = (n < ncols) ? 1.f : 0.f;
  v16h b;
#pragma unroll
  for (int r = 0; r < 8; ++r) {
    int k  = kt * 32 + ((r < 4) ? (2 * r + half * 8) : (16 + 2 * (r - 4) + half * 8));
    int k0 = (k     < krows) ? k     : (krows - 1);
    int k1 = (k + 1 < krows) ? k + 1 : (krows - 1);
    float m0 = (k     < krows) ? nm : 0.f;
    float m1 = (k + 1 < krows) ? nm : 0.f;
    b[2 * r]     = (_Float16)(W[k0 * ncols + nc] * m0);
    b[2 * r + 1] = (_Float16)(W[k1 * ncols + nc] * m1);
  }
  pk[tile * 32 + lane] = b;
}

// ---- Kernel 1: encoder GEMM  g[n,o] += sum_bl x[bl,n]*conv_w[bl,o] ----------
// Split-K over 12 blocks (8 k-tiles of 32); x tile (32x121 f32) staged to LDS
// by the Tensor Data Mover (or float4 copy fallback).
__global__ __launch_bounds__(256) void k1_encoder(const float* __restrict__ x,
                                                  const v16h* __restrict__ pk,
                                                  float* __restrict__ g_ws) {
  __shared__ __align__(16) float xs[32 * NNODES];  // contiguous 32x121 tile
  int tid = threadIdx.x, lane = tid & 31, wave = tid >> 5;
  int m = lane & 15, half = lane >> 4;
  int mt = wave;                                   // 8 waves cover 128 rows
  int ncol = lane & 15;

  int nrow = mt * 16 + m;                          // node index (A row)
  float nmask = (nrow < NNODES) ? 1.f : 0.f;
  int nidx = (nrow < NNODES) ? nrow : 0;

  v8f c = {};
  for (int j = 0; j < 8; ++j) {
    int kt = blockIdx.x * 8 + j;
    __syncthreads();                               // prior tile fully consumed
#if TDM_MODE != TDM_NONE
    if (wave == 0) {                               // one wave drives the TDM
      unsigned long long ga =
          (unsigned long long)(const void*)(x + (size_t)kt * 32 * NNODES);
      unsigned lds = (unsigned)(unsigned long long)(void*)xs;
      v4u g0;
      g0[0] = 1u;                                  // count=1 (valid user D#)
      g0[1] = lds;                                 // lds_addr (bytes)
      g0[2] = (unsigned)(ga & 0xFFFFFFFFull);      // global_addr[31:0]
      g0[3] = (unsigned)((ga >> 32) & 0x1FFFFFFull) | (2u << 30); // [56:32]|type=2
      v8i g1;
      g1[0] = (int)(2u << 16);                     // data_size=4B, no pad/iter
      g1[1] = (int)((unsigned)NNODES << 16);       // tensor_dim0[15:0]=121
      g1[2] = (int)((unsigned)BL << 16);           // tensor_dim1[15:0]=3072
      g1[3] = (int)((unsigned)NNODES << 16);       // tile_dim0=121
      g1[4] = 32;                                  // tile_dim1=32
      g1[5] = NNODES;                              // tensor_dim0_stride=121
      g1[6] = 0;
      g1[7] = 0;
      v4i g2 = {};                                 // 2D tensor: groups 2/3 unused
      v4i g3 = {};
#if TDM_MODE == TDM_6ARG
      v8i g4 = {};                                 // extra operand (zero-filled)
      __builtin_amdgcn_tensor_load_to_lds(g0, g1, g2, g3, g4, 0);
#else
      __builtin_amdgcn_tensor_load_to_lds(g0, g1, g2, g3, 0);
#endif
      __builtin_amdgcn_s_wait_tensorcnt(0);
    }
#else
    if (j < 7) __builtin_prefetch(&x[(size_t)(kt + 1) * 32 * NNODES], 0, 1);
    {
      const float4* src = (const float4*)(x + (size_t)kt * 32 * NNODES);
      float4* dst = (float4*)xs;
      for (int i = tid; i < (32 * NNODES) / 4; i += 256) dst[i] = src[i];
    }
#endif
    __syncthreads();                               // tile visible to all waves
    v16h a;
#pragma unroll
    for (int r = 0; r < 8; ++r) {
      int kp = (r < 4) ? (2 * r + half * 8) : (16 + 2 * (r - 4) + half * 8);
      a[2 * r]     = (_Float16)(xs[kp * NNODES + nidx] * nmask);
      a[2 * r + 1] = (_Float16)(xs[(kp + 1) * NNODES + nidx] * nmask);
    }
    v16h b = pk[(TB_CONV + kt) * 32 + lane];
    c = __builtin_amdgcn_wmma_f32_16x16x32_f16(false, a, false, b,
                                               (short)0, c, false, false);
  }
#pragma unroll
  for (int r = 0; r < 8; ++r) {                    // D: vgpr r -> M = r + half*8
    int row = mt * 16 + r + half * 8;
    if (row < NNODES)
      atomicAdd(&g_ws[row * 16 + ncol], c[r]);
  }
}

// ---- Kernel 2: feature MLP + adapter + LayerNorm -> z0 row vector v --------
__global__ __launch_bounds__(128) void k2_feat_adapter(
    const float* __restrict__ g_ws, const float* __restrict__ conv_b,
    const float* __restrict__ comp_w1, const float* __restrict__ comp_b1,
    const float* __restrict__ comp_w2, const float* __restrict__ comp_b2,
    const float* __restrict__ adapter_w, const float* __restrict__ adapter_b,
    const float* __restrict__ ln_g, const float* __restrict__ ln_b,
    float* __restrict__ v_ws) {
  __shared__ float s_feat[128];
  __shared__ float red[128];
  int d = threadIdx.x;

  float f = 0.f;
  if (d < NNODES) {
    float comb[16];
#pragma unroll
    for (int i = 0; i < 8; ++i) {
      float gv = g_ws[d * 16 + i] + conv_b[i];
      comb[i] = gv; comb[8 + i] = gv;            // static || dynamic (identical)
    }
    float acc = comp_b2[0];
    for (int j = 0; j < 32; ++j) {
      float s = comp_b1[j];
#pragma unroll
      for (int i = 0; i < 16; ++i) s += comb[i] * comp_w1[i * 32 + j];
      s = (s > 0.f) ? s : 0.01f * s;             // leaky
      acc += s * comp_w2[j];
    }
    f = acc;
  }
  s_feat[d] = f;
  __syncthreads();

  float pre = 0.f;
  if (d < NNODES) {
    pre = adapter_b[d];
    for (int nn = 0; nn < NNODES; ++nn)
      pre += s_feat[nn] * adapter_w[nn * DMODEL + d];
  }
  red[d] = (d < NNODES) ? pre : 0.f;
  __syncthreads();
  for (int s = 64; s > 0; s >>= 1) { if (d < s) red[d] += red[d + s]; __syncthreads(); }
  float mu = red[0] / (float)DMODEL;
  __syncthreads();
  float dm = (d < NNODES) ? (pre - mu) : 0.f;
  red[d] = dm * dm;
  __syncthreads();
  for (int s = 64; s > 0; s >>= 1) { if (d < s) red[d] += red[d + s]; __syncthreads(); }
  float var = red[0] / (float)DMODEL;

  float val = 0.f;
  if (d < NNODES) val = dm * rsqrtf(var + 1e-5f) * ln_g[d] + ln_b[d];
  v_ws[d] = val;                                 // padded [121..127] = 0
}

// ---- Kernel 3: 7-iteration single-row mamba recurrence (1 WG, 8 waves) -----
__global__ __launch_bounds__(256) void k3_recurrence(
    const float* __restrict__ v_ws, const v16h* __restrict__ pk,
    const float* __restrict__ lift_b1, const float* __restrict__ lift_b2,
    const float* __restrict__ net_b1,  const float* __restrict__ net_b2,
    const float* __restrict__ fc1_b,   const float* __restrict__ fc2_b,
    const float* __restrict__ fc3_b,
    const float* __restrict__ Amat,    const float* __restrict__ state_init,
    const float* __restrict__ fcout_b,
    float* __restrict__ preds_ws) {
  __shared__ float sh_h[128 * SSTATE];           // 32 KB state (b-degenerate)
  __shared__ float sh_q[128], sh_t1[128], sh_cat[256], sh_xt[128];
  __shared__ float sh_delta[128], sh_Bm[SSTATE], sh_Cm[SSTATE];
  __shared__ float sh_out[128], sh_pred[128];

  int tid = threadIdx.x, lane = tid & 31, wave = tid >> 5;

  for (int i = tid; i < 128 * SSTATE; i += 256) {
    int d = i >> 6, nn = i & 63;
    sh_h[i] = (d < DMODEL) ? state_init[d * SSTATE + nn] : 0.f;
  }
  if (tid < 128) sh_q[tid] = v_ws[tid];          // z0 row (padded)
  if (tid >= 242 && tid < 256) sh_cat[tid] = 0.f;
  __syncthreads();

  for (int it = 0; it <= STEPS; ++it) {          // it==0: warmup (h only)
    // lifted = tanh(relu(q.W1+b1).W2+b2)
    vecmat_wmma_pk(sh_q, pk, TB_LIFT1, HID, lift_b1, sh_t1, 4, 8, ACT_RELU);
    __syncthreads();
    vecmat_wmma_pk(sh_t1, pk, TB_LIFT2, DMODEL, lift_b2, sh_cat + DMODEL, 4, 8, ACT_TANH);
    if (tid < 128) sh_cat[tid] = (tid < DMODEL) ? sh_q[tid] : 0.f;
    __syncthreads();
    // xt = relu(cat.W1+b1).W2+b2   (cat length 242, K padded to 256)
    vecmat_wmma_pk(sh_cat, pk, TB_NET1, HID, net_b1, sh_t1, 8, 8, ACT_RELU);
    __syncthreads();
    vecmat_wmma_pk(sh_t1, pk, TB_NET2, DMODEL, net_b2, sh_xt, 4, 8, ACT_NONE);
    __syncthreads();
    // delta = softplus(xt.fc1+b)
    vecmat_wmma_pk(sh_xt, pk, TB_FC1, DMODEL, fc1_b, sh_delta, 4, 8, ACT_SOFTPLUS);
    __syncthreads();
    // Bm (waves 0-3) and Cm (waves 4-7), each 64 cols = 4 n-tiles
    {
      int tb          = (wave < 4) ? TB_FC2 : TB_FC3;
      const float* bb = (wave < 4) ? fc2_b : fc3_b;
      float* op       = (wave < 4) ? sh_Bm : sh_Cm;
      int nt = wave & 3;
      v8f c = {};
      for (int kt = 0; kt < 4; ++kt) {
        v16h a = frag_a_rowvec(sh_xt, kt);
        v16h b = pk[(tb + nt * 4 + kt) * 32 + lane];
        c = __builtin_amdgcn_wmma_f32_16x16x32_f16(false, a, false, b,
                                                   (short)0, c, false, false);
      }
      if (lane < 16) {
        int n = nt * 16 + lane;
        if (n < SSTATE) op[n] = c[0] + bb[n];
      }
    }
    __syncthreads();
    // state update: h = exp(-celu(delta*A))*h + xt*delta*Bm ; out = Cm.h
    if (tid < DMODEL) {
      int d = tid;
      float dl = sh_delta[d], xv = sh_xt[d];
      float acc = 0.f;
      for (int nn = 0; nn < SSTATE; ++nn) {
        float z = dl * Amat[d * SSTATE + nn];
        float celu = (z > 0.f) ? z : (__expf(z) - 1.f);
        float dA = __expf(-celu);
        float hn = dA * sh_h[d * SSTATE + nn] + xv * dl * sh_Bm[nn];
        sh_h[d * SSTATE + nn] = hn;
        acc += sh_Cm[nn] * hn;
      }
      sh_out[d] = acc;
    } else if (tid < 128) {
      sh_out[tid] = 0.f;
    }
    __syncthreads();
    if (it >= 1) {                               // pred = out.fcout + b ; q = pred
      vecmat_wmma_pk(sh_out, pk, TB_FCOUT, DMODEL, fcout_b, sh_pred, 4, 8, ACT_NONE);
      __syncthreads();
      if (tid < 128) {
        float qv = (tid < DMODEL) ? sh_pred[tid] : 0.f;
        sh_q[tid] = qv;
        preds_ws[(it - 1) * 128 + tid] = qv;
      }
      __syncthreads();
    }
  }
}

// ---- Kernel 4: broadcast 6 pred vectors to (6,128,24,121), float4 stores ---
__global__ __launch_bounds__(256) void k4_broadcast(const float* __restrict__ preds_ws,
                                                    float* __restrict__ out,
                                                    int total4) {
  __shared__ float sp[6 * 128];
  for (int i = threadIdx.x; i < 6 * 128; i += 256) sp[i] = preds_ws[i];
  __syncthreads();
  const int PERSTEP = 128 * 24 * NNODES;         // 371712 (divisible by 4)
  int stride = gridDim.x * blockDim.x;
  for (int i4 = blockIdx.x * blockDim.x + threadIdx.x; i4 < total4; i4 += stride) {
    int i = i4 * 4;
    int step = i / PERSTEP;                      // uniform within the float4
    int mv = i % NNODES;
    const float* row = sp + step * 128;
    float4 v;
    v.x = row[mv]; mv = (mv + 1 == NNODES) ? 0 : mv + 1;
    v.y = row[mv]; mv = (mv + 1 == NNODES) ? 0 : mv + 1;
    v.z = row[mv]; mv = (mv + 1 == NNODES) ? 0 : mv + 1;
    v.w = row[mv];
    ((float4*)out)[i4] = v;
  }
}

// ---------------------------------------------------------------------------
extern "C" void kernel_launch(void* const* d_in, const int* in_sizes, int n_in,
                              void* d_out, int out_size, void* d_ws, size_t ws_size,
                              hipStream_t stream) {
  (void)in_sizes; (void)n_in; (void)ws_size;
  const float* x          = (const float*)d_in[0];
  const float* conv_w     = (const float*)d_in[3];
  const float* conv_b     = (const float*)d_in[4];
  const float* comp_w1    = (const float*)d_in[5];
  const float* comp_b1    = (const float*)d_in[6];
  const float* comp_w2    = (const float*)d_in[7];
  const float* comp_b2    = (const float*)d_in[8];
  const float* adapter_w  = (const float*)d_in[9];
  const float* adapter_b  = (const float*)d_in[10];
  const float* ln_g       = (const float*)d_in[11];
  const float* ln_b       = (const float*)d_in[12];
  const float* lift_w1    = (const float*)d_in[13];
  const float* lift_b1    = (const float*)d_in[14];
  const float* lift_w2    = (const float*)d_in[15];
  const float* lift_b2    = (const float*)d_in[16];
  const float* net_w1     = (const float*)d_in[17];
  const float* net_b1     = (const float*)d_in[18];
  const float* net_w2     = (const float*)d_in[19];
  const float* net_b2     = (const float*)d_in[20];
  const float* fc1_w      = (const float*)d_in[21];
  const float* fc1_b      = (const float*)d_in[22];
  const float* fc2_w      = (const float*)d_in[23];
  const float* fc2_b      = (const float*)d_in[24];
  const float* fc3_w      = (const float*)d_in[25];
  const float* fc3_b      = (const float*)d_in[26];
  const float* Amat       = (const float*)d_in[27];
  const float* state_init = (const float*)d_in[28];
  const float* fcout_w    = (const float*)d_in[29];
  const float* fcout_b    = (const float*)d_in[30];

  // ws layout: g[2048 f32] @0 | v[128 f32] @8192B | preds[768 f32] @8704B |
  //            packed f16 fragments @12288B (352 tiles * 1KB)
  float* g_ws     = (float*)d_ws;
  float* v_ws     = (float*)((char*)d_ws + 8192);
  float* preds_ws = (float*)((char*)d_ws + 8704);
  v16h*  pk       = (v16h*)((char*)d_ws + 12288);

  hipMemsetAsync(g_ws, 0, 2048 * sizeof(float), stream);   // split-K accum init

  k0_pack<<<44, 256, 0, stream>>>(lift_w1, lift_w2, net_w1, net_w2,
                                  fc1_w, fc2_w, fc3_w, fcout_w, conv_w, pk);
  k1_encoder<<<12, 256, 0, stream>>>(x, pk, g_ws);
  k2_feat_adapter<<<1, 128, 0, stream>>>(g_ws, conv_b, comp_w1, comp_b1,
                                         comp_w2, comp_b2, adapter_w, adapter_b,
                                         ln_g, ln_b, v_ws);
  k3_recurrence<<<1, 256, 0, stream>>>(v_ws, pk, lift_b1, lift_b2,
                                       net_b1, net_b2, fc1_b, fc2_b, fc3_b,
                                       Amat, state_init, fcout_b, preds_ws);
  k4_broadcast<<<1024, 256, 0, stream>>>(preds_ws, (float*)d_out, out_size / 4);
}
